// ComprehensiveHierarchicalModel_38603166056954
// MI455X (gfx1250) — compile-verified, matrix operand order checked by reading, not run on hardware
//
#include <hip/hip_runtime.h>

typedef unsigned short u16;
typedef __attribute__((ext_vector_type(8)))  __bf16 bf8;
typedef __attribute__((ext_vector_type(16))) __bf16 bf16v;
typedef __attribute__((ext_vector_type(8)))  float  f32x8;

#define ACT_NONE 0
#define ACT_RELU 1
#define ACT_SIG  2

__device__ __forceinline__ float bf2f(u16 h) {
  union { unsigned u; float f; } v; v.u = ((unsigned)h) << 16; return v.f;
}
__device__ __forceinline__ u16 f2bf(float f) {
  union { float f; unsigned u; } v; v.f = f;
  unsigned r = ((v.u >> 16) & 1u) + 0x7fffu;
  return (u16)((v.u + r) >> 16);
}
__device__ __forceinline__ float sigmoidf(float x) { return 1.0f / (1.0f + expf(-x)); }

__device__ __forceinline__ bf16v combine8(bf8 lo, bf8 hi) {
  return __builtin_shufflevector(lo, hi, 0,1,2,3,4,5,6,7,8,9,10,11,12,13,14,15);
}

// ---------------------------------------------------------------------------
// Generic bf16 GEMM: Y[M,N] = act( A[M,K] @ W[N,ldb(>=K)]^T + bias + addend )
// Block = 256 threads = 8 waves arranged 2(M) x 4(N).
// Wave tile 32x64: 8 wmma accumulators (2 A-tiles x 4 B-tiles) ->
//   12 b128 loads per 8 WMMAs (1.5 loads/wmma).
// Block tile 64x256. Requires: M%64==0, N%256==0, K%32==0.
// ---------------------------------------------------------------------------
__global__ __launch_bounds__(256)
void gemm_bf16_kernel(const u16* __restrict__ A, const u16* __restrict__ W,
                      int M, int N, int K, int ldb,
                      const float* __restrict__ bias,
                      const float* __restrict__ addend,
                      float* __restrict__ outF, u16* __restrict__ outB,
                      int act)
{
  const int lane = threadIdx.x & 31;
  const int wid  = threadIdx.x >> 5;
  const int wm   = wid & 1;            // 2 waves along M
  const int wn   = wid >> 1;           // 4 waves along N
  const int row0 = blockIdx.y * 64 + wm * 32;
  const int n0   = blockIdx.x * 256 + wn * 64;
  const int r    = lane & 15;
  const int hs   = lane >> 4;          // K-half selector for 16-bit operand layout

  const u16* pA0 = A + (size_t)(row0 + r) * K + hs * 8;
  const u16* pA1 = pA0 + (size_t)16 * K;
  const u16* pB[4];
#pragma unroll
  for (int j = 0; j < 4; ++j)
    pB[j] = W + (size_t)(n0 + 16 * j + r) * ldb + hs * 8;

  // Warm first-touch pass (weights are L2-resident on later passes).
#pragma unroll
  for (int j = 0; j < 4; ++j) __builtin_prefetch(pB[j], 0, 3);

  f32x8 acc[8] = {};   // acc[m*4+j]

  for (int k = 0; k < K; k += 32) {
    bf16v a0 = combine8(*(const bf8*)(pA0 + k), *(const bf8*)(pA0 + k + 16));
    bf16v a1 = combine8(*(const bf8*)(pA1 + k), *(const bf8*)(pA1 + k + 16));
    bf16v b0 = combine8(*(const bf8*)(pB[0] + k), *(const bf8*)(pB[0] + k + 16));
    bf16v b1 = combine8(*(const bf8*)(pB[1] + k), *(const bf8*)(pB[1] + k + 16));
    bf16v b2 = combine8(*(const bf8*)(pB[2] + k), *(const bf8*)(pB[2] + k + 16));
    bf16v b3 = combine8(*(const bf8*)(pB[3] + k), *(const bf8*)(pB[3] + k + 16));
    acc[0] = __builtin_amdgcn_wmma_f32_16x16x32_bf16(false, a0, false, b0, (short)0, acc[0], false, false);
    acc[1] = __builtin_amdgcn_wmma_f32_16x16x32_bf16(false, a0, false, b1, (short)0, acc[1], false, false);
    acc[2] = __builtin_amdgcn_wmma_f32_16x16x32_bf16(false, a0, false, b2, (short)0, acc[2], false, false);
    acc[3] = __builtin_amdgcn_wmma_f32_16x16x32_bf16(false, a0, false, b3, (short)0, acc[3], false, false);
    acc[4] = __builtin_amdgcn_wmma_f32_16x16x32_bf16(false, a1, false, b0, (short)0, acc[4], false, false);
    acc[5] = __builtin_amdgcn_wmma_f32_16x16x32_bf16(false, a1, false, b1, (short)0, acc[5], false, false);
    acc[6] = __builtin_amdgcn_wmma_f32_16x16x32_bf16(false, a1, false, b2, (short)0, acc[6], false, false);
    acc[7] = __builtin_amdgcn_wmma_f32_16x16x32_bf16(false, a1, false, b3, (short)0, acc[7], false, false);
  }

  // Epilogue: D layout is lane l, vgpr rr -> m = rr + 8*(l>>4), n = l&15
  const int colLane = lane & 15;
#pragma unroll
  for (int m = 0; m < 2; ++m) {
    const int rowBase = row0 + 16 * m + 8 * hs;
#pragma unroll
    for (int j = 0; j < 4; ++j) {
      const int col = n0 + 16 * j + colLane;
      const float bv = bias ? bias[col] : 0.0f;
      const f32x8 a = acc[m * 4 + j];
#pragma unroll
      for (int rr = 0; rr < 8; ++rr) {
        const int rowi = rowBase + rr;
        const size_t idx = (size_t)rowi * N + col;
        float v = a[rr] + bv;
        if (addend) v += addend[idx];
        if (act == ACT_RELU)      v = fmaxf(v, 0.0f);
        else if (act == ACT_SIG)  v = sigmoidf(v);
        if (outF) outF[idx] = v;
        if (outB) outB[idx] = f2bf(v);
      }
    }
  }
}

// ---------------------------------------------------------------------------
// float -> bf16 cast (grid-stride)
// ---------------------------------------------------------------------------
__global__ void cast_f32_bf16_kernel(const float* __restrict__ src,
                                     u16* __restrict__ dst, size_t n)
{
  size_t i = (size_t)blockIdx.x * blockDim.x + threadIdx.x;
  size_t stride = (size_t)gridDim.x * blockDim.x;
  for (; i < n; i += stride) dst[i] = f2bf(src[i]);
}

// ---------------------------------------------------------------------------
// GRU gate combine: h' = (1-z)*tanh(i_n + r*h_n) + z*h ; gates sigmoid(i+h)
// gi, gh: [B, 3H]; h: [B, H]; dstF (+optional dstB): [B, H]
// ---------------------------------------------------------------------------
__global__ void gru_combine_kernel(const float* __restrict__ gi,
                                   const float* __restrict__ gh,
                                   const float* __restrict__ h,
                                   float* __restrict__ dstF,
                                   u16* __restrict__ dstB,
                                   int Bn, int H)
{
  size_t i = (size_t)blockIdx.x * blockDim.x + threadIdx.x;
  size_t n = (size_t)Bn * H;
  if (i >= n) return;
  int row = (int)(i / H);
  int c   = (int)(i % H);
  size_t base = (size_t)row * 3 * H + c;
  float ir = gi[base], iz = gi[base + H], in_ = gi[base + 2 * (size_t)H];
  float hr = gh[base], hz = gh[base + H], hn  = gh[base + 2 * (size_t)H];
  float rr = sigmoidf(ir + hr);
  float zz = sigmoidf(iz + hz);
  float nn = tanhf(in_ + rr * hn);
  float out = (1.0f - zz) * nn + zz * h[i];
  dstF[i] = out;
  if (dstB) dstB[i] = f2bf(out);
}

// ---------------------------------------------------------------------------
// LayerNorm (one block per row): y = (srcA[+srcB] - mu)*rstd*g + b [+ residual]
// If haltMask && haltMask[row], the row is skipped entirely (keeps old dst).
// ---------------------------------------------------------------------------
__global__ __launch_bounds__(256)
void ln_kernel(const float* __restrict__ srcA, const float* __restrict__ srcB,
               const float* __restrict__ gamma, const float* __restrict__ beta,
               const float* __restrict__ residual,
               const unsigned char* __restrict__ haltMask,
               float* __restrict__ dstF, u16* __restrict__ dstB, int H)
{
  const int row = blockIdx.x;
  if (haltMask && haltMask[row]) return;
  const int t = threadIdx.x;
  const int per = H >> 8;               // H/256, 4 or 8
  const float* a = srcA + (size_t)row * H;
  const float* b2 = srcB ? srcB + (size_t)row * H : nullptr;

  float vals[8];
  float s = 0.0f;
  for (int i = 0; i < per; ++i) {
    float v = a[t + i * 256];
    if (b2) v += b2[t + i * 256];
    vals[i] = v; s += v;
  }
  __shared__ float red[8];
  for (int off = 16; off; off >>= 1) s += __shfl_xor(s, off, 32);
  if ((t & 31) == 0) red[t >> 5] = s;
  __syncthreads();
  float tot = 0.0f;
  for (int i = 0; i < 8; ++i) tot += red[i];
  const float mean = tot / (float)H;
  __syncthreads();

  float vs = 0.0f;
  for (int i = 0; i < per; ++i) { float d = vals[i] - mean; vs += d * d; }
  for (int off = 16; off; off >>= 1) vs += __shfl_xor(vs, off, 32);
  if ((t & 31) == 0) red[t >> 5] = vs;
  __syncthreads();
  float vtot = 0.0f;
  for (int i = 0; i < 8; ++i) vtot += red[i];
  const float rstd = rsqrtf(vtot / (float)H + 1e-5f);

  for (int i = 0; i < per; ++i) {
    int c = t + i * 256;
    size_t idx = (size_t)row * H + c;
    float y = (vals[i] - mean) * rstd * gamma[c] + beta[c];
    if (residual) y += residual[idx];
    dstF[idx] = y;
    if (dstB) dstB[idx] = f2bf(y);
  }
}

// ---------------------------------------------------------------------------
// h_inp = bf16(agg_f32 * gate_f32)
// ---------------------------------------------------------------------------
__global__ void gate_apply_kernel(const float* __restrict__ aggF,
                                  const float* __restrict__ gateF,
                                  u16* __restrict__ dst, size_t n)
{
  size_t i = (size_t)blockIdx.x * blockDim.x + threadIdx.x;
  size_t stride = (size_t)gridDim.x * blockDim.x;
  for (; i < n; i += stride) dst[i] = f2bf(aggF[i] * gateF[i]);
}

// ---------------------------------------------------------------------------
// ACT halting head tail: p = sigmoid(dot(hidden_row, W2) + b2); update cum/halted
// One wave per row.
// ---------------------------------------------------------------------------
__global__ __launch_bounds__(256)
void halt_update_kernel(const u16* __restrict__ hidden,
                        const float* __restrict__ W2, const float* __restrict__ b2,
                        float* __restrict__ cum, unsigned char* __restrict__ halted,
                        int Bn, int Hh)
{
  int row = blockIdx.x * 8 + (threadIdx.x >> 5);
  int lane = threadIdx.x & 31;
  if (row >= Bn) return;
  float s = 0.0f;
  const u16* hr = hidden + (size_t)row * Hh;
  for (int c = lane; c < Hh; c += 32) s += bf2f(hr[c]) * W2[c];
  for (int off = 16; off; off >>= 1) s += __shfl_xor(s, off, 32);
  if (lane == 0) {
    if (!halted[row]) {
      float p = sigmoidf(s + b2[0]);
      float c2 = cum[row] + p;
      cum[row] = c2;
      if (c2 >= 0.99f) halted[row] = 1;
    }
  }
}

// ---------------------------------------------------------------------------
// Output head tail: out[row] = dot(hidden_row, W2) + b2   (OUT = 1)
// ---------------------------------------------------------------------------
__global__ __launch_bounds__(256)
void out_gemv_kernel(const u16* __restrict__ hidden,
                     const float* __restrict__ W2, const float* __restrict__ b2,
                     float* __restrict__ out, int Bn, int Hh)
{
  int row = blockIdx.x * 8 + (threadIdx.x >> 5);
  int lane = threadIdx.x & 31;
  if (row >= Bn) return;
  float s = 0.0f;
  const u16* hr = hidden + (size_t)row * Hh;
  for (int c = lane; c < Hh; c += 32) s += bf2f(hr[c]) * W2[c];
  for (int off = 16; off; off >>= 1) s += __shfl_xor(s, off, 32);
  if (lane == 0) out[row] = s + b2[0];
}

// ===========================================================================
// Host side
// ===========================================================================
extern "C" void kernel_launch(void* const* d_in, const int* in_sizes, int n_in,
                              void* d_out, int out_size, void* d_ws, size_t ws_size,
                              hipStream_t stream)
{
  constexpr int Bq = 4096, DIN = 2048, HL = 1024, HH = 2048, HHALT = 512;
  constexpr int CYCLES = 8, T = 3;

  // --- inputs (setup_inputs dict order) ---
  const float* x        = (const float*)d_in[0];
  const float* W_in     = (const float*)d_in[1];
  const float* b_in     = (const float*)d_in[2];
  const float* ln_in_g  = (const float*)d_in[3];
  const float* ln_in_b  = (const float*)d_in[4];
  const float* h2l_W    = (const float*)d_in[5];
  const float* h2l_b    = (const float*)d_in[6];
  const float* gruL_Wih = (const float*)d_in[7];
  const float* gruL_Whh = (const float*)d_in[8];
  const float* gruL_bih = (const float*)d_in[9];
  const float* gruL_bhh = (const float*)d_in[10];
  const float* lnL_g    = (const float*)d_in[11];
  const float* lnL_b    = (const float*)d_in[12];
  const float* agg_W    = (const float*)d_in[13];
  const float* agg_b    = (const float*)d_in[14];
  const float* gate_W   = (const float*)d_in[15];
  const float* gate_b   = (const float*)d_in[16];
  const float* gruH_Wih = (const float*)d_in[17];
  const float* gruH_Whh = (const float*)d_in[18];
  const float* gruH_bih = (const float*)d_in[19];
  const float* gruH_bhh = (const float*)d_in[20];
  const float* ln1_g    = (const float*)d_in[21];
  const float* ln1_b    = (const float*)d_in[22];
  const float* ffn_W1   = (const float*)d_in[23];
  const float* ffn_b1   = (const float*)d_in[24];
  const float* ffn_W2   = (const float*)d_in[25];
  const float* ffn_b2   = (const float*)d_in[26];
  const float* ln2_g    = (const float*)d_in[27];
  const float* ln2_b    = (const float*)d_in[28];
  const float* halt_W1  = (const float*)d_in[29];
  const float* halt_b1  = (const float*)d_in[30];
  const float* halt_W2  = (const float*)d_in[31];
  const float* halt_b2  = (const float*)d_in[32];
  const float* out_W1   = (const float*)d_in[33];
  const float* out_b1   = (const float*)d_in[34];
  const float* out_W2   = (const float*)d_in[35];
  const float* out_b2   = (const float*)d_in[36];
  (void)in_sizes; (void)n_in; (void)out_size; (void)ws_size;

  // --- workspace layout ---
  size_t off = 0;
  auto take = [&](size_t bytes) -> char* {
    char* p = (char*)d_ws + off;
    off += (bytes + 255) & ~(size_t)255;
    return p;
  };
  // bf16 weights (all fit in 192MB L2 once hot)
  u16* wb_in   = (u16*)take((size_t)HL * DIN * 2);
  u16* wb_h2l  = (u16*)take((size_t)HL * HH * 2);
  u16* wb_gLi  = (u16*)take((size_t)3 * HL * HL * 2);
  u16* wb_gLh  = (u16*)take((size_t)3 * HL * HL * 2);
  u16* wb_agg  = (u16*)take((size_t)HH * HL * 2);
  u16* wb_gate = (u16*)take((size_t)HH * 2 * HH * 2);
  u16* wb_gHi  = (u16*)take((size_t)3 * HH * HH * 2);
  u16* wb_gHh  = (u16*)take((size_t)3 * HH * HH * 2);
  u16* wb_f1   = (u16*)take((size_t)2 * HH * HH * 2);
  u16* wb_f2   = (u16*)take((size_t)HH * 2 * HH * 2);
  u16* wb_h1   = (u16*)take((size_t)HHALT * HH * 2);
  u16* wb_o1   = (u16*)take((size_t)HL * HH * 2);
  // activations
  u16*   x_b    = (u16*)  take((size_t)Bq * DIN * 2);
  float* xpre   = (float*)take((size_t)Bq * HL * 4);
  float* xemb_f = (float*)take((size_t)Bq * HL * 4);
  u16*   xemb_b = (u16*)  take((size_t)Bq * HL * 2);
  float* zL_f   = (float*)take((size_t)Bq * HL * 4);
  u16*   zL_b   = (u16*)  take((size_t)Bq * HL * 2);
  float* zH_f   = (float*)take((size_t)Bq * HH * 4);
  u16*   zH_b   = (u16*)  take((size_t)Bq * HH * 2);
  u16*   ctx_b  = (u16*)  take((size_t)Bq * HL * 2);
  float* agg_f  = (float*)take((size_t)Bq * HH * 4);
  u16*   agg_b16= (u16*)  take((size_t)Bq * HH * 2);
  float* gate_f = (float*)take((size_t)Bq * HH * 4);
  u16*   hinp_b = (u16*)  take((size_t)Bq * HH * 2);
  float* hgru_f = (float*)take((size_t)Bq * HH * 4);
  float* hnew_f = (float*)take((size_t)Bq * HH * 4);
  u16*   hnew_b = (u16*)  take((size_t)Bq * HH * 2);
  u16*   f1_b   = (u16*)  take((size_t)Bq * 2 * HH * 2);
  float* f2_f   = (float*)take((size_t)Bq * HH * 4);   // also aliased as gate_tmp
  u16*   hh_b   = (u16*)  take((size_t)Bq * HHALT * 2);
  u16*   oh_b   = (u16*)  take((size_t)Bq * HL * 2);
  float* cum    = (float*)take((size_t)Bq * 4);
  unsigned char* halted = (unsigned char*)take((size_t)Bq);
  // big GEMM-output region, reused across low/high phases
  char* big = take((size_t)Bq * 3 * HH * 4 * 2);       // 192 MB
  float* giL = (float*)big;
  float* ghL = (float*)(big + (size_t)Bq * 3 * HL * 4);
  float* giH = (float*)big;
  float* ghH = (float*)(big + (size_t)Bq * 3 * HH * 4);
  float* gate_tmp = f2_f;

  // --- helpers ---
  auto castw = [&](const float* src, u16* dst, size_t n) {
    cast_f32_bf16_kernel<<<2048, 256, 0, stream>>>(src, dst, n);
  };
  auto gemm = [&](const u16* A, const u16* W, int N, int K, int ldb,
                  const float* bias, const float* addend,
                  float* outF, u16* outB, int act) {
    dim3 g(N / 256, Bq / 64);
    gemm_bf16_kernel<<<g, 256, 0, stream>>>(A, W, Bq, N, K, ldb,
                                            bias, addend, outF, outB, act);
  };
  auto ln = [&](const float* sa, const float* sb, const float* g, const float* b,
                const float* res, const unsigned char* hm,
                float* dF, u16* dB, int H) {
    ln_kernel<<<Bq, 256, 0, stream>>>(sa, sb, g, b, res, hm, dF, dB, H);
  };
  auto gru = [&](const float* gi, const float* gh, const float* h,
                 float* dF, u16* dB, int H) {
    size_t n = (size_t)Bq * H;
    gru_combine_kernel<<<(unsigned)((n + 255) / 256), 256, 0, stream>>>(
        gi, gh, h, dF, dB, Bq, H);
  };

  // --- weight conversion (deterministic; done every call) ---
  castw(x, x_b, (size_t)Bq * DIN);
  castw(W_in,     wb_in,   (size_t)HL * DIN);
  castw(h2l_W,    wb_h2l,  (size_t)HL * HH);
  castw(gruL_Wih, wb_gLi,  (size_t)3 * HL * HL);
  castw(gruL_Whh, wb_gLh,  (size_t)3 * HL * HL);
  castw(agg_W,    wb_agg,  (size_t)HH * HL);
  castw(gate_W,   wb_gate, (size_t)HH * 2 * HH);
  castw(gruH_Wih, wb_gHi,  (size_t)3 * HH * HH);
  castw(gruH_Whh, wb_gHh,  (size_t)3 * HH * HH);
  castw(ffn_W1,   wb_f1,   (size_t)2 * HH * HH);
  castw(ffn_W2,   wb_f2,   (size_t)HH * 2 * HH);
  castw(halt_W1,  wb_h1,   (size_t)HHALT * HH);
  castw(out_W1,   wb_o1,   (size_t)HL * HH);

  // --- zero recurrent state ---
  hipMemsetAsync(zL_f,   0, (size_t)Bq * HL * 4, stream);
  hipMemsetAsync(zL_b,   0, (size_t)Bq * HL * 2, stream);
  hipMemsetAsync(zH_f,   0, (size_t)Bq * HH * 4, stream);
  hipMemsetAsync(zH_b,   0, (size_t)Bq * HH * 2, stream);
  hipMemsetAsync(cum,    0, (size_t)Bq * 4, stream);
  hipMemsetAsync(halted, 0, (size_t)Bq, stream);

  // --- input network: x_emb = LN(relu(x @ W_in^T + b_in)) ---
  gemm(x_b, wb_in, HL, DIN, DIN, b_in, nullptr, xpre, nullptr, ACT_RELU);
  ln(xpre, nullptr, ln_in_g, ln_in_b, nullptr, nullptr, xemb_f, xemb_b, HL);

  for (int cycle = 0; cycle < CYCLES; ++cycle) {
    // ctx = x_emb + z_H @ h2l^T + b   (bf16 only; GEMM input)
    gemm(zH_b, wb_h2l, HL, HH, HH, h2l_b, xemb_f, nullptr, ctx_b, ACT_NONE);
    // gi_L = ctx @ Wih^T + bih  (ctx fixed across T steps -> computed once)
    gemm(ctx_b, wb_gLi, 3 * HL, HL, HL, gruL_bih, nullptr, giL, nullptr, ACT_NONE);
    for (int t = 0; t < T; ++t) {
      gemm(zL_b, wb_gLh, 3 * HL, HL, HL, gruL_bhh, nullptr, ghL, nullptr, ACT_NONE);
      gru(giL, ghL, zL_f, zL_f, zL_b, HL);
    }
    ln(zL_f, nullptr, lnL_g, lnL_b, nullptr, nullptr, zL_f, zL_b, HL);

    // agg = relu(z_L @ agg_W^T + b)
    gemm(zL_b, wb_agg, HH, HL, HL, agg_b, nullptr, agg_f, agg_b16, ACT_RELU);
    // gate = sigmoid([z_H, agg] @ gate_W^T + b), split over gate_W column halves
    gemm(zH_b, wb_gate, HH, HH, 2 * HH, nullptr, nullptr, gate_tmp, nullptr, ACT_NONE);
    gemm(agg_b16, wb_gate + HH, HH, HH, 2 * HH, gate_b, gate_tmp, gate_f, nullptr, ACT_SIG);
    {
      size_t n = (size_t)Bq * HH;
      gate_apply_kernel<<<2048, 256, 0, stream>>>(agg_f, gate_f, hinp_b, n);
    }
    // high-level GRU
    gemm(hinp_b, wb_gHi, 3 * HH, HH, HH, gruH_bih, nullptr, giH, nullptr, ACT_NONE);
    gemm(zH_b,   wb_gHh, 3 * HH, HH, HH, gruH_bhh, nullptr, ghH, nullptr, ACT_NONE);
    gru(giH, ghH, zH_f, hgru_f, nullptr, HH);
    // h_new = LN(gru) + z_H
    ln(hgru_f, nullptr, ln1_g, ln1_b, zH_f, nullptr, hnew_f, hnew_b, HH);
    // FFN
    gemm(hnew_b, wb_f1, 2 * HH, HH, HH, ffn_b1, nullptr, nullptr, f1_b, ACT_RELU);
    gemm(f1_b,   wb_f2, HH, 2 * HH, 2 * HH, ffn_b2, nullptr, f2_f, nullptr, ACT_NONE);
    // z_H = halted ? z_H : LN(h_new + ffn)   (halted rows skipped in-place)
    ln(hnew_f, f2_f, ln2_g, ln2_b, nullptr, halted, zH_f, zH_b, HH);
    // ACT halting
    gemm(zH_b, wb_h1, HHALT, HH, HH, halt_b1, nullptr, nullptr, hh_b, ACT_RELU);
    halt_update_kernel<<<Bq / 8, 256, 0, stream>>>(hh_b, halt_W2, halt_b2,
                                                   cum, halted, Bq, HHALT);
  }

  // output head
  gemm(zH_b, wb_o1, HL, HH, HH, out_b1, nullptr, nullptr, oh_b, ACT_RELU);
  out_gemv_kernel<<<Bq / 8, 256, 0, stream>>>(oh_b, out_W2, out_b2,
                                              (float*)d_out, Bq, HL);
}
